// ResidualPCEncoder_19722489823712
// MI455X (gfx1250) — matrix operand order repaired and checked by reading, not compiled, not run on hardware
//
#include <hip/hip_runtime.h>
#include <hip/hip_bf16.h>

// ---------------------------------------------------------------------------
// ResidualPCEncoder forward for MI455X (gfx1250, wave32, WMMA).
// Heavy GEMMs use v_wmma_f32_16x16x32_f16 with a 128x32 block tile
// (two accumulators per wave, one A-register load per k-chunk).
// B panel staged TRANSPOSED in LDS so each lane fetches its 16 operand
// halves as one contiguous 32B LDS access (2x ds_load_b128) instead of
// 16 scalar reads. Edge-conv folded into two GEMMs + gather-max; kNN Gram
// matrix is WMMA-tiled with top-16 selection out of 128KB LDS per wave.
// ---------------------------------------------------------------------------

typedef __attribute__((ext_vector_type(16))) _Float16 v16h;
typedef __attribute__((ext_vector_type(8)))  _Float16 v8h;
typedef __attribute__((ext_vector_type(8)))  float    v8f;

#define NPTS   2048
#define BATCH  8
#define ROWS   16384        // B*N
#define HDIM   256
#define KNN    16
#define EPSBN  1e-5f

static __device__ __forceinline__ v8f wmma_f16(v16h a, v16h b, v8f c) {
  // (neg_a, A, neg_b, B, c_mod, C, reuse_a, reuse_b)
  return __builtin_amdgcn_wmma_f32_16x16x32_f16(false, a, false, b, (short)0, c,
                                                false, false);
}

// Build the 16-bit A-operand (16x32) register image for one
// lane: elems 0..15 -> K = kc*32 + 16*half + e (contiguous, matches B).
static __device__ __forceinline__ v16h load_a16(const _Float16* row, int kc, int half) {
  v8h lo = *(const v8h*)(row + kc * 32 + 16 * half);
  v8h hi = *(const v8h*)(row + kc * 32 + 16 * half + 8);
  v16h a;
#pragma unroll
  for (int e = 0; e < 8; ++e) { a[e] = lo[e]; a[e + 8] = hi[e]; }
  return a;
}

// ---------------------------------------------------------------------------
// f32 -> f16 weight conversion
// ---------------------------------------------------------------------------
__global__ void cvt_f16_kernel(const float* __restrict__ s, _Float16* __restrict__ d, int n) {
  int i = blockIdx.x * 256 + threadIdx.x;
  if (i < n) d[i] = (_Float16)s[i];
}

// ---------------------------------------------------------------------------
// h = x @ Wf + bf   (K=3, trivial), also emit f16 copy for the first GEMM
// ---------------------------------------------------------------------------
__global__ void featurize_kernel(const float* __restrict__ x, const float* __restrict__ Wf,
                                 const float* __restrict__ bf,
                                 float* __restrict__ h, _Float16* __restrict__ hh) {
  int n = blockIdx.x;            // 0..16383
  int c = threadIdx.x;           // 0..255
  float v = bf[c];
#pragma unroll
  for (int d = 0; d < 3; ++d) v += x[n * 3 + d] * Wf[d * HDIM + c];
  size_t o = (size_t)n * HDIM + c;
  h[o]  = v;
  hh[o] = (_Float16)v;
}

// ---------------------------------------------------------------------------
// WMMA GEMM: C[M=16384, N] = A[f16, M x 256] * B[f16, 256 x N] (+bias)
// MODE 0: C = acc + bias                        (bias may be null)
// MODE 1: C = resid + alpha[ai]*(acc + bias)    (residual update, resid==C ok)
//
// Block: 256 thr = 8 waves; block tile 128 rows x 32 cols. Full-K B panel
// (256 x 32) staged transposed in LDS: Bs[n][k], row stride padded to 264
// halves (528B) so the 16 lanes of a half-group hit different banks. Each
// wave computes two 16x16 column tiles reusing one A register load per kc.
// ---------------------------------------------------------------------------
template <int MODE>
__global__ void gemm_kernel(const _Float16* __restrict__ A, const _Float16* __restrict__ Bw,
                            const float* __restrict__ bias, float* __restrict__ Cout,
                            const float* __restrict__ resid, const float* __restrict__ alpha_p,
                            int ai, int N) {
  constexpr int BSTR = 264;                 // padded k-stride (halves)
  __shared__ _Float16 Bs[32 * BSTR];        // [n][k] transposed panel, ~16.5 KB

  const int tid  = threadIdx.x;
  const int n0   = blockIdx.x * 32;
  const int row0 = blockIdx.y * 128;

  // Stage B panel transposed: thread tid owns global k-row tid; vector-load
  // 32 contiguous f16 (64B) then scatter into LDS columns.
  {
    union { uint4 q[4]; _Float16 hh[32]; } u;
    const uint4* src = (const uint4*)(Bw + (size_t)tid * N + n0);
    u.q[0] = src[0]; u.q[1] = src[1]; u.q[2] = src[2]; u.q[3] = src[3];
#pragma unroll
    for (int j = 0; j < 32; ++j) Bs[j * BSTR + tid] = u.hh[j];
  }
  __syncthreads();

  const int w    = tid >> 5;
  const int lane = tid & 31;
  const int mloc = lane & 15;
  const int half = lane >> 4;

  const _Float16* arow = A + (size_t)(row0 + w * 16 + mloc) * HDIM;

  v8f acc0 = {};
  v8f acc1 = {};
#pragma unroll
  for (int kc = 0; kc < 8; ++kc) {
    if (kc < 7) __builtin_prefetch(arow + (kc + 1) * 32, 0, 0);  // global_prefetch_b8
    v16h a = load_a16(arow, kc, half);
    // B operand: lane's column n, 16 contiguous k-halves -> 32B LDS load.
    v16h b0 = *(const v16h*)(&Bs[(mloc)      * BSTR + kc * 32 + 16 * half]);
    v16h b1 = *(const v16h*)(&Bs[(mloc + 16) * BSTR + kc * 32 + 16 * half]);
    acc0 = wmma_f16(a, b0, acc0);
    acc1 = wmma_f16(a, b1, acc1);
  }

  const float al = (MODE == 1) ? alpha_p[ai] : 0.0f;
#pragma unroll
  for (int g = 0; g < 2; ++g) {
    const int n = n0 + g * 16 + mloc;
    const float bb = bias ? bias[n] : 0.0f;
    const v8f& acc = g ? acc1 : acc0;
#pragma unroll
    for (int r = 0; r < 8; ++r) {
      int m = row0 + w * 16 + r + 8 * half;  // C layout: VGPR r -> M=r (+8 hi half)
      size_t o = (size_t)m * N + n;
      float v = acc[r] + bb;
      if (MODE == 1) v = resid[o] + al * v;
      Cout[o] = v;
    }
  }
}

// ---------------------------------------------------------------------------
// BatchNorm statistics: per-channel mean & inv-std over 16384 rows
// ---------------------------------------------------------------------------
__global__ void bn_stats_kernel(const float* __restrict__ x, int C,
                                float* __restrict__ mean, float* __restrict__ inv) {
  __shared__ float sh[256], sh2[256];
  int c = blockIdx.x, t = threadIdx.x;
  float s = 0.f, s2 = 0.f;
  for (int n = t; n < ROWS; n += 256) {
    float v = x[(size_t)n * C + c];
    s += v; s2 += v * v;
  }
  sh[t] = s; sh2[t] = s2;
  __syncthreads();
  for (int o = 128; o > 0; o >>= 1) {
    if (t < o) { sh[t] += sh[t + o]; sh2[t] += sh2[t + o]; }
    __syncthreads();
  }
  if (t == 0) {
    float m = sh[0] * (1.0f / ROWS);
    float v = sh2[0] * (1.0f / ROWS) - m * m;
    mean[c] = m;
    inv[c]  = rsqrtf(v + EPSBN);
  }
}

// y = relu((x-mean)*inv*g + b); optional f32 and f16 outputs (in-place safe)
__global__ void bn_apply_kernel(const float* __restrict__ x, const float* __restrict__ mean,
                                const float* __restrict__ inv, const float* __restrict__ g,
                                const float* __restrict__ b, int C,
                                float* __restrict__ outf, _Float16* __restrict__ outh) {
  size_t i = (size_t)blockIdx.x * 256 + threadIdx.x;
  int c = (int)(i % C);
  float y = (x[i] - mean[c]) * inv[c] * g[c] + b[c];
  y = fmaxf(y, 0.0f);
  if (outf) outf[i] = y;
  if (outh) outh[i] = (_Float16)y;
}

// ---------------------------------------------------------------------------
// Per-row squared norm (wave32 per row)
// ---------------------------------------------------------------------------
__global__ void sqnorm_kernel(const float* __restrict__ t, float* __restrict__ sq) {
  int wave = threadIdx.x >> 5, lane = threadIdx.x & 31;
  int row = blockIdx.x * 8 + wave;           // grid 2048
  float s = 0.f;
  for (int c = lane; c < HDIM; c += 32) { float v = t[(size_t)row * HDIM + c]; s += v * v; }
  for (int o = 16; o >= 1; o >>= 1) s += __shfl_xor(s, o, 32);
  if (lane == 0) sq[row] = s;
}

// ---------------------------------------------------------------------------
// kNN: one wave per 16 query rows. Gram tiles via WMMA (both operands load
// directly in WMMA lane layout: B columns are candidate feature rows, which
// are contiguous in memory). Distances to 128KB dynamic LDS, then 16x argmin
// extraction per row with wave32 shuffle reductions.
// ---------------------------------------------------------------------------
__global__ void knn_kernel(const _Float16* __restrict__ th, const float* __restrict__ sq,
                           int* __restrict__ idx_out) {
  extern __shared__ float dist[];            // [16][2048] = 128 KB

  const int blk  = blockIdx.x;               // 0..1023
  const int b    = blk >> 7;                 // batch
  const int rt   = blk & 127;                // row tile in batch
  const int lane = threadIdx.x;              // single wave32
  const int mloc = lane & 15;
  const int half = lane >> 4;

  const int qbase = b * NPTS + rt * 16;
  const _Float16* arow = th + (size_t)(qbase + mloc) * HDIM;

  v16h areg[8];
#pragma unroll
  for (int kc = 0; kc < 8; ++kc) areg[kc] = load_a16(arow, kc, half);

  float sqm[8];
#pragma unroll
  for (int r = 0; r < 8; ++r) sqm[r] = sq[qbase + r + 8 * half];

  for (int mt = 0; mt < 128; ++mt) {
    const int p = b * NPTS + mt * 16 + mloc; // candidate point for this lane's column
    const float sqc = sq[p];
    const _Float16* crow = th + (size_t)p * HDIM;
    v8f acc = {};
#pragma unroll
    for (int kc = 0; kc < 8; ++kc) {
      // B operand column n = candidate point; K rows 16*half+e are contiguous.
      v16h bb = *(const v16h*)(crow + kc * 32 + half * 16);
      acc = wmma_f16(areg[kc], bb, acc);
    }
#pragma unroll
    for (int r = 0; r < 8; ++r) {
      dist[(r + 8 * half) * NPTS + mt * 16 + mloc] = sqm[r] + sqc - 2.0f * acc[r];
    }
  }
  __syncthreads();

  for (int row = 0; row < 16; ++row) {
    float* drow = dist + row * NPTS;
    const int grow = qbase + row;
    for (int k = 0; k < KNN; ++k) {
      float best = 3.0e38f; int bi = 0;
      for (int j = lane; j < NPTS; j += 32) {
        float v = drow[j];
        if (v < best) { best = v; bi = j; }
      }
      for (int o = 16; o >= 1; o >>= 1) {
        float ov = __shfl_xor(best, o, 32);
        int   oi = __shfl_xor(bi, o, 32);
        if (ov < best || (ov == best && oi < bi)) { best = ov; bi = oi; }
      }
      if (lane == 0) {
        idx_out[(size_t)grow * KNN + k] = b * NPTS + bi;  // global row index
        drow[bi] = 3.0e38f;
      }
      __syncthreads();
    }
  }
}

// ---------------------------------------------------------------------------
// Edge-conv fold: out[n] = A[n] - Bv[n] + max_k Bv[idx[n][k]]
// (A already includes bias be from its GEMM epilogue)
// ---------------------------------------------------------------------------
__global__ void gather_max_kernel(const float* __restrict__ Aa, const float* __restrict__ Bv,
                                  const int* __restrict__ idx, float* __restrict__ out) {
  __shared__ int id[KNN];
  int n = blockIdx.x, c = threadIdx.x;
  if (c < KNN) id[c] = idx[(size_t)n * KNN + c];
  __syncthreads();
  size_t o = (size_t)n * HDIM + c;
  float base = Aa[o] - Bv[o];
  float m = -3.0e38f;
#pragma unroll
  for (int k = 0; k < KNN; ++k) m = fmaxf(m, Bv[(size_t)id[k] * HDIM + c]);
  out[o] = base + m;
}

// ---------------------------------------------------------------------------
// Global max pool over points per batch: p[b][g] = max_n t3[b,n,g]
// ---------------------------------------------------------------------------
__global__ void batch_max_kernel(const float* __restrict__ t3, float* __restrict__ p) {
  int b = blockIdx.x, g = threadIdx.x;       // 8 x 512
  float m = -3.0e38f;
  for (int n = 0; n < NPTS; ++n)
    m = fmaxf(m, t3[((size_t)(b * NPTS + n)) * 512 + g]);
  p[b * 512 + g] = m;
}

// ---------------------------------------------------------------------------
// Tiny head MLP: (8x512) -> relu 256 -> relu 128 -> 128 (fp32, one block)
// ---------------------------------------------------------------------------
__global__ void head_mlp_kernel(const float* __restrict__ p,
                                const float* __restrict__ Wm1, const float* __restrict__ bm1,
                                const float* __restrict__ Wm2, const float* __restrict__ bm2,
                                const float* __restrict__ Wm3, const float* __restrict__ bm3,
                                float* __restrict__ out) {
  __shared__ float p1[8 * 256];
  __shared__ float p2[8 * 128];
  int t = threadIdx.x;
  for (int e = t; e < 8 * 256; e += 256) {
    int r = e >> 8, c = e & 255;
    float s = bm1[c];
    for (int k = 0; k < 512; ++k) s += p[r * 512 + k] * Wm1[k * 256 + c];
    p1[e] = fmaxf(s, 0.0f);
  }
  __syncthreads();
  for (int e = t; e < 8 * 128; e += 256) {
    int r = e >> 7, c = e & 127;
    float s = bm2[c];
    for (int k = 0; k < 256; ++k) s += p1[r * 256 + k] * Wm2[k * 128 + c];
    p2[e] = fmaxf(s, 0.0f);
  }
  __syncthreads();
  for (int e = t; e < 8 * 128; e += 256) {
    int r = e >> 7, c = e & 127;
    float s = bm3[c];
    for (int k = 0; k < 128; ++k) s += p2[r * 128 + k] * Wm3[k * 128 + c];
    out[e] = s;
  }
}

// ---------------------------------------------------------------------------
// Host-side orchestration
// ---------------------------------------------------------------------------
extern "C" void kernel_launch(void* const* d_in, const int* in_sizes, int n_in,
                              void* d_out, int out_size, void* d_ws, size_t ws_size,
                              hipStream_t stream) {
  (void)in_sizes; (void)n_in; (void)out_size; (void)ws_size;

  const float* x    = (const float*)d_in[0];
  // d_in[1] = batch (unused)
  const float* Wf   = (const float*)d_in[2];
  const float* bf   = (const float*)d_in[3];
  const float* Wnn  = (const float*)d_in[4];
  const float* bnn  = (const float*)d_in[5];
  const float* g1   = (const float*)d_in[6];
  const float* b1   = (const float*)d_in[7];
  const float* We   = (const float*)d_in[8];
  const float* be   = (const float*)d_in[9];
  const float* g2   = (const float*)d_in[10];
  const float* b2   = (const float*)d_in[11];
  const float* Wl   = (const float*)d_in[12];
  const float* bl   = (const float*)d_in[13];
  const float* alpha= (const float*)d_in[14];
  const float* gg   = (const float*)d_in[15];
  const float* bgb  = (const float*)d_in[16];
  const float* Wg   = (const float*)d_in[17];
  const float* bg2  = (const float*)d_in[18];
  const float* Wm1  = (const float*)d_in[19];
  const float* bm1  = (const float*)d_in[20];
  const float* Wm2  = (const float*)d_in[21];
  const float* bm2  = (const float*)d_in[22];
  const float* Wm3  = (const float*)d_in[23];
  const float* bm3  = (const float*)d_in[24];

  char* ws = (char*)d_ws;
  constexpr size_t MB = 1ull << 20;
  float*     h    = (float*)(ws + 0 * MB);                 // 16 MB
  float*     t    = (float*)(ws + 16 * MB);                // 16 MB
  float*     tA   = (float*)(ws + 32 * MB);                // 16 MB (t3 aliases 32MB..64MB)
  float*     tB   = (float*)(ws + 48 * MB);                // 16 MB
  _Float16*  th   = (_Float16*)(ws + 64 * MB);             // 8 MB
  _Float16*  hh   = (_Float16*)(ws + 72 * MB);             // 8 MB
  int*       kidx = (int*)(ws + 80 * MB);                  // 1 MB
  float*     sq   = (float*)(ws + 81 * MB);                // 64 KB
  float*     mean = (float*)(ws + 81 * MB + (64 << 10));
  float*     inv  = (float*)(ws + 81 * MB + (68 << 10));
  _Float16*  wWnn = (_Float16*)(ws + 82 * MB);             // 128 KB
  _Float16*  wWe  = (_Float16*)(ws + 82 * MB + (256 << 10));// 1 MB
  _Float16*  wWl  = (_Float16*)(ws + 84 * MB);             // 512 KB
  _Float16*  wWg  = (_Float16*)(ws + 85 * MB);             // 256 KB
  float*     pbuf = (float*)(ws + 86 * MB);                // 16 KB
  float*     t3   = tA;                                    // 16384 x 512 f32

  // Weight conversion to f16 (WMMA operand precision)
  cvt_f16_kernel<<<(65536 + 255) / 256, 256, 0, stream>>>(Wnn, wWnn, 65536);
  cvt_f16_kernel<<<(524288 + 255) / 256, 256, 0, stream>>>(We, wWe, 524288);
  cvt_f16_kernel<<<(262144 + 255) / 256, 256, 0, stream>>>(Wl, wWl, 262144);
  cvt_f16_kernel<<<(131072 + 255) / 256, 256, 0, stream>>>(Wg, wWg, 131072);

  // h = x @ Wf + bf
  featurize_kernel<<<ROWS, 256, 0, stream>>>(x, Wf, bf, h, hh);

  const dim3 gemm_block(256);
  const dim3 gemm_grid256(HDIM / 32, ROWS / 128);   // N=256 -> 8 x 128
  const dim3 gemm_grid512(512 / 32, ROWS / 128);    // N=512 -> 16 x 128

  for (int i = 0; i < 4; ++i) {
    const float* pre;
    if (i == 0) {
      gemm_kernel<0><<<gemm_grid256, gemm_block, 0, stream>>>(
          hh, wWnn, bnn, t, nullptr, nullptr, 0, HDIM);
      pre = t;
    } else {
      pre = h;
    }
    // BN1 + ReLU -> t (f32) and th (f16)
    bn_stats_kernel<<<HDIM, 256, 0, stream>>>(pre, HDIM, mean, inv);
    bn_apply_kernel<<<ROWS, 256, 0, stream>>>(pre, mean, inv, g1 + i * HDIM, b1 + i * HDIM,
                                              HDIM, t, th);
    // kNN in 256-d feature space (WMMA Gram + LDS top-16)
    sqnorm_kernel<<<NPTS, 256, 0, stream>>>(t, sq);
    knn_kernel<<<BATCH * (NPTS / 16), 32, 16 * NPTS * sizeof(float), stream>>>(th, sq, kidx);
    // Edge conv folded: A = t@We_top + be ; Bv = t@We_bot
    gemm_kernel<0><<<gemm_grid256, gemm_block, 0, stream>>>(
        th, wWe + (size_t)i * 512 * HDIM, be + i * HDIM, tA, nullptr, nullptr, 0, HDIM);
    gemm_kernel<0><<<gemm_grid256, gemm_block, 0, stream>>>(
        th, wWe + (size_t)i * 512 * HDIM + (size_t)HDIM * HDIM, nullptr, tB, nullptr, nullptr, 0, HDIM);
    gather_max_kernel<<<ROWS, 256, 0, stream>>>(tA, tB, kidx, t);
    // BN2 + ReLU
    bn_stats_kernel<<<HDIM, 256, 0, stream>>>(t, HDIM, mean, inv);
    bn_apply_kernel<<<ROWS, 256, 0, stream>>>(t, mean, inv, g2 + i * HDIM, b2 + i * HDIM,
                                              HDIM, t, th);
    // h = h + alpha[i] * (t @ Wl[i] + bl[i])
    gemm_kernel<1><<<gemm_grid256, gemm_block, 0, stream>>>(
        th, wWl + (size_t)i * HDIM * HDIM, bl + i * HDIM, h, h, alpha, i, HDIM);
  }

  // Final: relu(bn(h)) @ Wg + bg2 -> max over points -> head MLP
  bn_stats_kernel<<<HDIM, 256, 0, stream>>>(h, HDIM, mean, inv);
  bn_apply_kernel<<<ROWS, 256, 0, stream>>>(h, mean, inv, gg, bgb, HDIM, nullptr, th);
  gemm_kernel<0><<<gemm_grid512, gemm_block, 0, stream>>>(
      th, wWg, bg2, t3, nullptr, nullptr, 0, 512);
  batch_max_kernel<<<BATCH, 512, 0, stream>>>(t3, pbuf);
  head_mlp_kernel<<<1, 256, 0, stream>>>(pbuf, Wm1, bm1, Wm2, bm2, Wm3, bm3, (float*)d_out);
}
